// MoEST_Plus_81922206204350
// MI455X (gfx1250) — matrix operand order repaired and checked by reading, not compiled, not run on hardware
//
#include <hip/hip_runtime.h>
#include <hip/hip_bf16.h>
#include <math.h>
#include <stdint.h>

// ---------------------------------------------------------------------------
// MoEST_Plus forward for MI455X (gfx1250, wave32, WMMA).
// - All GEMMs: v_wmma_f32_16x16x32_f16 (f16 inputs, fp32 accumulate).
// - Activations feeding GEMM A-operands are kept in f16 in global memory
//   (numerically identical to converting at stage time, half the traffic).
// - Flash attention; Q/K tiles staged with GLOBAL_LOAD_ASYNC_TO_LDS_B128.
// ---------------------------------------------------------------------------

typedef __attribute__((ext_vector_type(16))) _Float16 v16h;
typedef __attribute__((ext_vector_type(8)))  _Float16 v8h;
typedef __attribute__((ext_vector_type(8)))  float    v8f;

__device__ inline v8f wmma_f16(v16h a, v16h b, v8f c) {
  // 8 args: (neg_a, A, neg_b, B, c_mod, C, reuse_a, reuse_b)
  return __builtin_amdgcn_wmma_f32_16x16x32_f16(false, a, false, b, (short)0, c,
                                                false, false);
}

// A fragment (16x32 MxK, f16) from a row-major LDS tile (stride ld elems).
// CDNA5 ISA 7.12.2: lane L -> row M = L&15, halfsel hs = L>>4;
// slots 0..7 <- K = 8*hs+0..7 ; slots 8..15 <- K = 16+8*hs+0..7.
__device__ inline v16h frag_a(const _Float16* base, int ld, int lane) {
  int r = lane & 15, hs = (lane >> 4) & 1;
  const _Float16* p = base + r * ld + 8 * hs;
  v8h lo = *(const v8h*)(p);
  v8h hi = *(const v8h*)(p + 16);
  v16h f;
#pragma unroll
  for (int i = 0; i < 8; i++) { f[i] = lo[i]; f[i + 8] = hi[i]; }
  return f;
}

// B fragment (32x16 KxN, f16) from an N-major (transposed) LDS tile [n][k].
// lane L -> col N = L&15 ; slots 0..7 <- K = 8*hs+0..7 ; slots 8..15 <- K = 16+8*hs+0..7.
__device__ inline v16h frag_b(const _Float16* base, int ld, int lane) {
  int r = lane & 15, hs = (lane >> 4) & 1;
  const _Float16* p = base + r * ld + 8 * hs;
  v8h lo = *(const v8h*)(p);
  v8h hi = *(const v8h*)(p + 16);
  v16h f;
#pragma unroll
  for (int i = 0; i < 8; i++) { f[i] = lo[i]; f[i + 8] = hi[i]; }
  return f;
}

__device__ inline float gelu_exact(float x) {
  return 0.5f * x * (1.f + erff(x * 0.70710678118654752f));
}
__device__ inline float softplus_f(float x) {
  return fmaxf(x, 0.f) + log1pf(__expf(-fabsf(x)));
}

// --- CDNA5 async global->LDS copy (ASYNCcnt), ISA §10.7 / §15.18.3 ---------
__device__ inline unsigned lds_off(void* p) {
  // generic -> LDS(addrspace 3) -> 32-bit wave-relative LDS byte offset
  return (unsigned)(unsigned long long)(__attribute__((address_space(3))) void*)p;
}
__device__ inline void async_copy_b128(unsigned lds_addr, const void* gaddr) {
  unsigned long long ga = (unsigned long long)(uintptr_t)gaddr;
  asm volatile("global_load_async_to_lds_b128 %0, %1, off"
               :: "v"(lds_addr), "v"(ga) : "memory");
}
__device__ inline void async_wait0() {
  asm volatile("s_wait_asynccnt 0x0" ::: "memory");
}

// ---------------------------------------------------------------------------
// Generic WMMA GEMM: out = epilogue(A[MxK] * W[KxN] + bias)
// Block tile 128x128, BK=32; 8 waves, each wave a 32x64 sub-tile (8 wmma/step).
// A16: A operand is f16 in global (K must be a multiple of 32); else fp32.
// ---------------------------------------------------------------------------
enum { EP_PLAIN = 0, EP_H16 = 1, EP_GELU16 = 2, EP_MOE = 3, EP_NBD = 4 };

template <int MODE, bool A16>
__global__ __launch_bounds__(256) void wgemm_kernel(
    const void* __restrict__ Ap, const float* __restrict__ W,
    const float* __restrict__ bias, const float* __restrict__ res,
    float* __restrict__ out, float* __restrict__ out2,
    _Float16* __restrict__ outh,
    const float* __restrict__ gate, int expert, int Gnum,
    int M, int N, int K) {
  __shared__ alignas(16) _Float16 As[128][40];  // [m][k], pad 8 (16B rows)
  __shared__ alignas(16) _Float16 Bs[128][40];  // [n][k] (W transposed on stage)

  const int tid = threadIdx.x;
  const int m0 = blockIdx.y * 128;
  const int n0 = blockIdx.x * 128;
  const int w = tid >> 5, lane = tid & 31;
  const int wm = w >> 1, wn = w & 1;
  const int hs = lane >> 4, ln16 = lane & 15;

  v8f acc[2][4];
#pragma unroll
  for (int f = 0; f < 2; f++)
#pragma unroll
    for (int g = 0; g < 4; g++)
#pragma unroll
      for (int i = 0; i < 8; i++) acc[f][g][i] = 0.f;

  for (int kb = 0; kb < K; kb += 32) {
    __syncthreads();
    if (A16) {  // stage A (128x32) f16 -> f16, 16B vector moves
      const _Float16* Ah = (const _Float16*)Ap;
      int r0 = tid >> 2;          // 0..63
      int c0 = (tid & 3) << 3;    // 0,8,16,24
#pragma unroll
      for (int p = 0; p < 2; p++) {
        int r = r0 + p * 64;
        int gm = m0 + r;
        v8h val;
        if (gm < M) {
          val = *(const v8h*)&Ah[(long)gm * K + kb + c0];
        } else {
#pragma unroll
          for (int j = 0; j < 8; j++) val[j] = (_Float16)0.f;
        }
        *(v8h*)&As[r][c0] = val;
      }
      if (tid < 32 && kb + 32 < K && (m0 + tid * 4) < M)
        __builtin_prefetch(&Ah[(long)(m0 + tid * 4) * K + kb + 32], 0, 1);
    } else {    // stage A (128x32) fp32 -> f16
      const float* Af = (const float*)Ap;
      int r0 = tid >> 3;          // 0..31
      int c0 = (tid & 7) << 2;    // 0..28
#pragma unroll
      for (int p = 0; p < 4; p++) {
        int r = r0 + p * 32;
        int gm = m0 + r;
#pragma unroll
        for (int j = 0; j < 4; j++) {
          int gk = kb + c0 + j;
          float v = (gm < M && gk < K) ? Af[(long)gm * K + gk] : 0.f;
          As[r][c0 + j] = (_Float16)v;
        }
      }
      if (tid < 32 && kb + 32 < K && (m0 + tid * 4) < M)
        __builtin_prefetch(&Af[(long)(m0 + tid * 4) * K + kb + 32], 0, 1);
    }
    {  // stage B transposed: Bs[n][k] <- W[k][n], fp32 -> f16
      int n4 = (tid & 31) << 2;   // 0..124
      int kr = tid >> 5;          // 0..7
#pragma unroll
      for (int p = 0; p < 4; p++) {
        int kk = kr + p * 8;
        int gk = kb + kk;
#pragma unroll
        for (int j = 0; j < 4; j++) {
          int gn = n0 + n4 + j;
          float v = (gk < K && gn < N) ? W[(long)gk * N + gn] : 0.f;
          Bs[n4 + j][kk] = (_Float16)v;
        }
      }
    }
    __syncthreads();

    v16h af[2], bf[4];
#pragma unroll
    for (int f = 0; f < 2; f++) af[f] = frag_a(&As[wm * 32 + f * 16][0], 40, lane);
#pragma unroll
    for (int g = 0; g < 4; g++) bf[g] = frag_b(&Bs[wn * 64 + g * 16][0], 40, lane);
#pragma unroll
    for (int f = 0; f < 2; f++)
#pragma unroll
      for (int g = 0; g < 4; g++) acc[f][g] = wmma_f16(af[f], bf[g], acc[f][g]);
  }

  // epilogue — C layout: row = f*16 + hs*8 + r, col = g*16 + ln16
#pragma unroll
  for (int f = 0; f < 2; f++)
#pragma unroll
    for (int g = 0; g < 4; g++)
#pragma unroll
      for (int r = 0; r < 8; r++) {
        int row = m0 + wm * 32 + f * 16 + hs * 8 + r;
        int col = n0 + wn * 64 + g * 16 + ln16;
        if (row < M && col < N) {
          float v = acc[f][g][r];
          if (bias) v += bias[col];
          if (MODE == EP_PLAIN) {
            if (res) v += res[(long)row * N + col];
            out[(long)row * N + col] = v;
            if (outh) outh[(long)row * N + col] = (_Float16)v;
          } else if (MODE == EP_H16) {
            outh[(long)row * N + col] = (_Float16)v;
          } else if (MODE == EP_GELU16) {
            outh[(long)row * N + col] = (_Float16)gelu_exact(v);
          } else if (MODE == EP_MOE) {
            out[(long)row * N + col] += gate[row * 4 + expert] * v;
          } else {  // EP_NBD: col = 2*g + comp; softplus split into mu/theta
            int gi = col >> 1;
            float sp = softplus_f(v);
            if ((col & 1) == 0) out[(long)row * Gnum + gi] = sp;
            else out2[(long)row * Gnum + gi] = sp + 1e-6f;
          }
        }
      }
}

// ---------------------------------------------------------------------------
// Flash attention: one block = (head, 128 query rows). HD=64, H=4, D=256.
// q/k/v/ctx are f16 [Ntok][256]. Q/K tiles staged via async global->LDS.
// Per 32-key step & wave: 4 wmma (S = Q*K^T) + 4 wmma (P*V).
// 1/sqrt(64) scale folded into the softmax step.
// ---------------------------------------------------------------------------
__global__ __launch_bounds__(256) void attn_kernel(
    const _Float16* __restrict__ q, const _Float16* __restrict__ k,
    const _Float16* __restrict__ v, _Float16* __restrict__ ctx, int Ntok) {
  __shared__ alignas(16) _Float16 Qs[128][72];    // [row][dim]
  __shared__ alignas(16) _Float16 Ks[32][72];     // [key][dim] == B^T layout
  __shared__ alignas(16) _Float16 Vt[64][40];     // [dim][key] == B^T layout
  __shared__ alignas(16) _Float16 Ps[8][16][40];  // per-wave P (16x32)

  const int tid = threadIdx.x;
  const int head = blockIdx.y;
  const int q0 = blockIdx.x * 128;
  const int w = tid >> 5, lane = tid & 31;
  const int hs = lane >> 4, ln16 = lane & 15;
  const int qr = w * 16;

  // stage Q (128 rows x 64 dims = 16KB) with async b128 copies
  for (int idx = tid; idx < 128 * 8; idx += 256) {
    int r = idx >> 3, c8 = (idx & 7) << 3;
    async_copy_b128(lds_off(&Qs[r][c8]),
                    &q[(long)(q0 + r) * 256 + head * 64 + c8]);
  }

  v8f o0, o1, o2, o3;
#pragma unroll
  for (int i = 0; i < 8; i++) { o0[i] = 0.f; o1[i] = 0.f; o2[i] = 0.f; o3[i] = 0.f; }
  float m_i[8], l_i[8];
#pragma unroll
  for (int r = 0; r < 8; r++) { m_i[r] = -1e30f; l_i[r] = 0.f; }

  const int nkt = Ntok >> 5;
  for (int kt = 0; kt < nkt; ++kt) {
    const int k0 = kt * 32;
    __syncthreads();
    {  // K tile (32x64) async copy: one b128 per thread
      int r = tid >> 3, c8 = (tid & 7) << 3;
      async_copy_b128(lds_off(&Ks[r][c8]),
                      &k[(long)(k0 + r) * 256 + head * 64 + c8]);
    }
    {  // V tile transposed [dim][key]: v8h load + 8 b16 stores
      int key = tid >> 3, c8 = (tid & 7) << 3;
      v8h val = *(const v8h*)&v[(long)(k0 + key) * 256 + head * 64 + c8];
#pragma unroll
      for (int j = 0; j < 8; j++) Vt[c8 + j][key] = val[j];
    }
    async_wait0();
    __syncthreads();

    // S = Q(16x64) * K^T(64x32): split K-dim into two 32-chunks
    v16h aq0 = frag_a(&Qs[qr][0], 72, lane);
    v16h aq1 = frag_a(&Qs[qr][32], 72, lane);
    v8f s0, s1;
#pragma unroll
    for (int i = 0; i < 8; i++) { s0[i] = 0.f; s1[i] = 0.f; }
    s0 = wmma_f16(aq0, frag_b(&Ks[0][0], 72, lane), s0);
    s0 = wmma_f16(aq1, frag_b(&Ks[0][32], 72, lane), s0);
    s1 = wmma_f16(aq0, frag_b(&Ks[16][0], 72, lane), s1);
    s1 = wmma_f16(aq1, frag_b(&Ks[16][32], 72, lane), s1);

    // online softmax across the 32 keys of this step (scale = 1/8)
    float corr[8];
#pragma unroll
    for (int r = 0; r < 8; r++) {
      float sc0 = s0[r] * 0.125f, sc1 = s1[r] * 0.125f;
      float mx = fmaxf(sc0, sc1);
#pragma unroll
      for (int off = 1; off < 16; off <<= 1) mx = fmaxf(mx, __shfl_xor(mx, off, 32));
      float mn = fmaxf(m_i[r], mx);
      float c0 = __expf(m_i[r] - mn);
      float p0 = __expf(sc0 - mn);
      float p1 = __expf(sc1 - mn);
      float rs = p0 + p1;
#pragma unroll
      for (int off = 1; off < 16; off <<= 1) rs += __shfl_xor(rs, off, 32);
      l_i[r] = l_i[r] * c0 + rs;
      m_i[r] = mn;
      corr[r] = c0;
      Ps[w][hs * 8 + r][ln16] = (_Float16)p0;
      Ps[w][hs * 8 + r][16 + ln16] = (_Float16)p1;
    }
#pragma unroll
    for (int r = 0; r < 8; r++) {
      o0[r] *= corr[r]; o1[r] *= corr[r]; o2[r] *= corr[r]; o3[r] *= corr[r];
    }
    __syncthreads();  // make per-wave P stores visible before fragment loads

    // O += P(16x32) * V(32x64)
    v16h ap = frag_a(&Ps[w][0][0], 40, lane);
    o0 = wmma_f16(ap, frag_b(&Vt[0][0], 40, lane), o0);
    o1 = wmma_f16(ap, frag_b(&Vt[16][0], 40, lane), o1);
    o2 = wmma_f16(ap, frag_b(&Vt[32][0], 40, lane), o2);
    o3 = wmma_f16(ap, frag_b(&Vt[48][0], 40, lane), o3);
  }

#pragma unroll
  for (int r = 0; r < 8; r++) {
    float inv = 1.f / l_i[r];
    int row = q0 + qr + hs * 8 + r;
    _Float16* outp = &ctx[(long)row * 256 + head * 64];
    outp[ln16]      = (_Float16)(o0[r] * inv);
    outp[16 + ln16] = (_Float16)(o1[r] * inv);
    outp[32 + ln16] = (_Float16)(o2[r] * inv);
    outp[48 + ln16] = (_Float16)(o3[r] * inv);
  }
}

// ---------------------------------------------------------------------------
// Small kernels
// ---------------------------------------------------------------------------
__global__ void fourier_kernel(const float* __restrict__ pos,
                               const float* __restrict__ Bf,
                               _Float16* __restrict__ F, int Ntok) {
  int idx = blockIdx.x * blockDim.x + threadIdx.x;
  if (idx >= Ntok * 64) return;
  int n = idx >> 6, m = idx & 63;
  float xp = 6.28318530717958648f *
             (pos[n * 3 + 0] * Bf[m] + pos[n * 3 + 1] * Bf[64 + m] +
              pos[n * 3 + 2] * Bf[128 + m]);
  F[n * 128 + m] = (_Float16)sinf(xp);
  F[n * 128 + 64 + m] = (_Float16)cosf(xp);
}

template <bool GELU>
__global__ __launch_bounds__(256) void ln_kernel(float* __restrict__ x,
                                                 _Float16* __restrict__ xh,
                                                 const float* __restrict__ g,
                                                 const float* __restrict__ b,
                                                 int M) {
  int row = blockIdx.x * 8 + (threadIdx.x >> 5);
  int lane = threadIdx.x & 31;
  if (row >= M) return;
  float v[8], s = 0.f, s2 = 0.f;
#pragma unroll
  for (int i = 0; i < 8; i++) {
    v[i] = x[(long)row * 256 + i * 32 + lane];
    s += v[i]; s2 += v[i] * v[i];
  }
#pragma unroll
  for (int off = 1; off < 32; off <<= 1) {
    s += __shfl_xor(s, off, 32);
    s2 += __shfl_xor(s2, off, 32);
  }
  float mean = s * (1.f / 256.f);
  float var = s2 * (1.f / 256.f) - mean * mean;
  float rstd = rsqrtf(var + 1e-5f);
#pragma unroll
  for (int i = 0; i < 8; i++) {
    int c = i * 32 + lane;
    float y = (v[i] - mean) * rstd * g[c] + b[c];
    if (GELU) y = gelu_exact(y);
    x[(long)row * 256 + c] = y;
    if (xh) xh[(long)row * 256 + c] = (_Float16)y;
  }
}

__global__ void h16_copy_kernel(const float* __restrict__ src,
                                _Float16* __restrict__ dst, long n) {
  long i = (long)blockIdx.x * blockDim.x + threadIdx.x;
  if (i < n) dst[i] = (_Float16)src[i];
}

__global__ void router_kernel(const float* __restrict__ zn,
                              const float* __restrict__ grad,
                              const float* __restrict__ Wr,
                              const float* __restrict__ br,
                              float* __restrict__ gate, int Ntok) {
  int n = blockIdx.x * blockDim.x + threadIdx.x;
  if (n >= Ntok) return;
  float a0 = br[0], a1 = br[1], a2 = br[2], a3 = br[3];
  for (int j = 0; j < 256; j++) {
    float zj = zn[(long)n * 256 + j];
    a0 += zj * Wr[j * 4 + 0]; a1 += zj * Wr[j * 4 + 1];
    a2 += zj * Wr[j * 4 + 2]; a3 += zj * Wr[j * 4 + 3];
  }
  float gr = grad[n];
  a0 += gr * Wr[1024 + 0]; a1 += gr * Wr[1024 + 1];
  a2 += gr * Wr[1024 + 2]; a3 += gr * Wr[1024 + 3];
  float lg[4] = {a0, a1, a2, a3};
  float mx = fmaxf(fmaxf(a0, a1), fmaxf(a2, a3));
  float pe[4], s = 0.f;
  for (int e = 0; e < 4; e++) { pe[e] = __expf(lg[e] - mx); s += pe[e]; }
  int am = 0;
  for (int e = 1; e < 4; e++) if (pe[e] > pe[am]) am = e;  // first-max tie rule
  for (int e = 0; e < 4; e++) gate[n * 4 + e] = (e == am) ? pe[e] / s : 0.f;
}

__global__ void ghead_kernel(const _Float16* __restrict__ f1,
                             const float* __restrict__ Wf2,
                             const float* __restrict__ bf2,
                             float* __restrict__ gout, int Ntok) {
  int n = blockIdx.x * blockDim.x + threadIdx.x;
  if (n >= Ntok) return;
  float a = bf2[0];
  for (int i = 0; i < 64; i++) a += (float)f1[(long)n * 64 + i] * Wf2[i];
  gout[n] = 1.f / (1.f + __expf(-a));
}

// ---------------------------------------------------------------------------
// Orchestration
// ---------------------------------------------------------------------------
extern "C" void kernel_launch(void* const* d_in, const int* in_sizes, int n_in,
                              void* d_out, int out_size, void* d_ws, size_t ws_size,
                              hipStream_t stream) {
  const float* vis = (const float*)d_in[0];
  const float* pos = (const float*)d_in[1];
  const float* grad = (const float*)d_in[2];
  const float* Bf = (const float*)d_in[3];
  const float* W_img = (const float*)d_in[4];
  const float* b_img = (const float*)d_in[5];
  const float* W_pos = (const float*)d_in[6];
  const float* b_pos = (const float*)d_in[7];
  const float* Wq = (const float*)d_in[8];  const float* bq = (const float*)d_in[9];
  const float* Wk = (const float*)d_in[10]; const float* bk = (const float*)d_in[11];
  const float* Wv = (const float*)d_in[12]; const float* bv = (const float*)d_in[13];
  const float* Wo = (const float*)d_in[14]; const float* bo = (const float*)d_in[15];
  const float* ln1_g = (const float*)d_in[16]; const float* ln1_b = (const float*)d_in[17];
  const float* W_router = (const float*)d_in[18]; const float* b_router = (const float*)d_in[19];
  const float* W1e = (const float*)d_in[20]; const float* b1e = (const float*)d_in[21];
  const float* W2e = (const float*)d_in[22]; const float* b2e = (const float*)d_in[23];
  const float* Wg1 = (const float*)d_in[24]; const float* bg1 = (const float*)d_in[25];
  const float* lng_g = (const float*)d_in[26]; const float* lng_b = (const float*)d_in[27];
  const float* Wg2 = (const float*)d_in[28]; const float* bg2 = (const float*)d_in[29];
  const float* Wf1 = (const float*)d_in[30]; const float* bf1 = (const float*)d_in[31];
  const float* Wf2 = (const float*)d_in[32]; const float* bf2 = (const float*)d_in[33];

  const int Ntok = in_sizes[1] / 3;       // 4096
  const int DU = in_sizes[0] / Ntok;      // 1024
  const int Gn = in_sizes[29] / 2;        // 2000
  const long NC = (long)Ntok * 256;

  // fp32 region
  float* Z = (float*)d_ws;
  float* ZN = Z + NC;
  float* ZMOE = ZN + NC;
  float* ENC2 = ZMOE + NC;
  float* GATE = ENC2 + NC;                // Ntok*4
  // f16 region (16B aligned: all offsets are multiples of 4 floats)
  _Float16* Qh = (_Float16*)(GATE + (long)Ntok * 4);
  _Float16* Kh = Qh + NC;
  _Float16* Vh = Kh + NC;
  _Float16* CTXh = Vh + NC;
  _Float16* Zh = CTXh + NC;
  _Float16* ZNh = Zh + NC;
  _Float16* ZMOEh = ZNh + NC;
  _Float16* Fh = ZMOEh + NC;              // Ntok*128
  _Float16* Hh = Qh;                      // alias: Ntok*1024 == 4*NC halves
  _Float16* T1h = Zh;                     // alias (Zh dead after q/k/v GEMMs)
  _Float16* F1h = Fh;                     // alias (Fh dead after enc2 GEMM)
  float* T1 = Z;                          // alias (Z dead after proj GEMM)

  float* out = (float*)d_out;
  float* mu = out;
  float* theta = out + (long)Ntok * Gn;
  float* gout = out + 2L * Ntok * Gn;

  dim3 blk(256);
  const int MB = (Ntok + 127) / 128;

  // 1) fourier features (f16)
  fourier_kernel<<<(Ntok * 64 + 255) / 256, blk, 0, stream>>>(pos, Bf, Fh, Ntok);
  // 2) enc2 = fourier @ W_pos + b_pos (fp32)
  wgemm_kernel<EP_PLAIN, true><<<dim3(2, MB), blk, 0, stream>>>(
      Fh, W_pos, b_pos, nullptr, ENC2, nullptr, nullptr, nullptr, 0, 0,
      Ntok, 256, 128);
  // 3) z = vis @ W_img + b_img + enc2 (fp32 + f16 mirror)
  wgemm_kernel<EP_PLAIN, false><<<dim3(2, MB), blk, 0, stream>>>(
      vis, W_img, b_img, ENC2, Z, nullptr, Zh, nullptr, 0, 0, Ntok, 256, DU);
  // 4) q,k,v projections (f16 out)
  wgemm_kernel<EP_H16, true><<<dim3(2, MB), blk, 0, stream>>>(
      Zh, Wq, bq, nullptr, nullptr, nullptr, Qh, nullptr, 0, 0, Ntok, 256, 256);
  wgemm_kernel<EP_H16, true><<<dim3(2, MB), blk, 0, stream>>>(
      Zh, Wk, bk, nullptr, nullptr, nullptr, Kh, nullptr, 0, 0, Ntok, 256, 256);
  wgemm_kernel<EP_H16, true><<<dim3(2, MB), blk, 0, stream>>>(
      Zh, Wv, bv, nullptr, nullptr, nullptr, Vh, nullptr, 0, 0, Ntok, 256, 256);
  // 5) flash attention (4 heads), f16 in/out
  attn_kernel<<<dim3(Ntok / 128, 4), blk, 0, stream>>>(Qh, Kh, Vh, CTXh, Ntok);
  // 6) zn = ctx @ Wo + bo + z ; LN1 (fp32 + f16 mirror)
  wgemm_kernel<EP_PLAIN, true><<<dim3(2, MB), blk, 0, stream>>>(
      CTXh, Wo, bo, Z, ZN, nullptr, nullptr, nullptr, 0, 0, Ntok, 256, 256);
  ln_kernel<false><<<(Ntok + 7) / 8, blk, 0, stream>>>(ZN, ZNh, ln1_g, ln1_b, Ntok);
  // 7) router (top-1 gate)
  router_kernel<<<(Ntok + 255) / 256, blk, 0, stream>>>(
      ZN, grad, W_router, b_router, GATE, Ntok);
  // 8) MoE: zmoe = zn + sum_e gate[:,e]*(gelu(zn@W1e[e]+b1e[e])@W2e[e]+b2e[e])
  hipMemcpyAsync(ZMOE, ZN, NC * sizeof(float), hipMemcpyDeviceToDevice, stream);
  for (int e = 0; e < 4; e++) {
    wgemm_kernel<EP_GELU16, true><<<dim3(8, MB), blk, 0, stream>>>(
        ZNh, W1e + (long)e * 256 * 1024, b1e + (long)e * 1024, nullptr,
        nullptr, nullptr, Hh, nullptr, 0, 0, Ntok, 1024, 256);
    wgemm_kernel<EP_MOE, true><<<dim3(2, MB), blk, 0, stream>>>(
        Hh, W2e + (long)e * 1024 * 256, b2e + (long)e * 256, nullptr, ZMOE,
        nullptr, nullptr, GATE, e, 0, Ntok, 256, 1024);
  }
  h16_copy_kernel<<<(unsigned)((NC + 255) / 256), blk, 0, stream>>>(ZMOE, ZMOEh, NC);
  // 9) gene decoder: t = gelu(LN(zmoe@Wg1+bg1)); preds -> softplus mu/theta
  wgemm_kernel<EP_PLAIN, true><<<dim3(2, MB), blk, 0, stream>>>(
      ZMOEh, Wg1, bg1, nullptr, T1, nullptr, nullptr, nullptr, 0, 0,
      Ntok, 256, 256);
  ln_kernel<true><<<(Ntok + 7) / 8, blk, 0, stream>>>(T1, T1h, lng_g, lng_b, Ntok);
  wgemm_kernel<EP_NBD, true><<<dim3((2 * Gn + 127) / 128, MB), blk, 0, stream>>>(
      T1h, Wg2, bg2, nullptr, mu, theta, nullptr, nullptr, 0, Gn,
      Ntok, 2 * Gn, 256);
  // 10) functional head
  wgemm_kernel<EP_GELU16, true><<<dim3(1, MB), blk, 0, stream>>>(
      ZMOEh, Wf1, bf1, nullptr, nullptr, nullptr, F1h, nullptr, 0, 0,
      Ntok, 64, 256);
  ghead_kernel<<<(Ntok + 255) / 256, blk, 0, stream>>>(F1h, Wf2, bf2, gout, Ntok);

  (void)n_in; (void)out_size; (void)ws_size;
}
